// MultHeadAttention_48292612276260
// MI455X (gfx1250) — compile-verified, matrix operand order checked by reading, not run on hardware
//
#include <hip/hip_runtime.h>

// ---------------- problem constants ----------------
#define EMB   1024
#define NH    16
#define HDIM  64
#define BB    4
#define SS    2048
#define M1    (BB*SS)        // 8192 rows (B*S)
#define N1    (3*EMB)        // 3072
#define KDIM  EMB            // 1024
#define QSZ   (BB*NH*SS*HDIM) // 8388608 elements per q/k/v tensor

typedef __bf16 bf16_t;
typedef __attribute__((ext_vector_type(16))) __bf16 v16bf;
typedef __attribute__((ext_vector_type(8)))  __bf16 v8bf;
typedef __attribute__((ext_vector_type(4)))  __bf16 v4bf;
typedef __attribute__((ext_vector_type(8)))  float  v8f;
typedef __attribute__((ext_vector_type(4)))  float  v4f;
typedef __attribute__((ext_vector_type(4)))  unsigned int u32x4;
typedef __attribute__((ext_vector_type(8)))  int  i32x8;
typedef __attribute__((ext_vector_type(4)))  int  i32x4;

// ---------------- WMMA helpers ----------------
__device__ __forceinline__ v8f wmma_bf16(v16bf a, v16bf b, v8f c) {
  // D = A(16x32) x B(32x16) + C, f32 accumulate
  return __builtin_amdgcn_wmma_f32_16x16x32_bf16(false, a, false, b, (short)0, c, false, false);
}

// Load one 16x32 bf16 fragment (A or B layout) given a pointer to the start of
// this lane's 32-element K-run.  Per ISA 7.12.2: lane half h selects K offset
// 8h for slots 0..7 and 16+8h for slots 8..15.
__device__ __forceinline__ v16bf frag32(const bf16_t* p, int h) {
  v8bf lo = *(const v8bf*)(p + 8 * h);
  v8bf hi = *(const v8bf*)(p + 16 + 8 * h);
  return __builtin_shufflevector(lo, hi, 0,1,2,3,4,5,6,7,8,9,10,11,12,13,14,15);
}

// ---------------- TDM: 2D bf16 tile -> LDS, with row padding ----------------
// Builds D# per CDNA5 ISA and issues TENSOR_LOAD_TO_LDS.
// tile_d0 contiguous elements per row, tile_d1 rows, row stride stride0
// (all in 2-byte elements).  Pad engine: after every 64 B stored, add 16 B
// (pad_interval=3, pad_amount=3)  ==>  LDS row stride 80 B = 40 bf16 = ASTR.
__device__ __forceinline__ void tdm_load_tile_bf16(
    unsigned lds_off, unsigned long long gaddr,
    unsigned tile_d0, unsigned tile_d1,
    unsigned long long tensor_d0, unsigned long long tensor_d1,
    unsigned long long stride0) {
  u32x4 g0;
  g0[0] = 1u;                                              // count=1 (valid)
  g0[1] = lds_off;                                         // lds_addr (bytes)
  g0[2] = (unsigned)gaddr;                                 // global_addr[31:0]
  g0[3] = ((unsigned)(gaddr >> 32) & 0x01FFFFFFu)          // global_addr[56:32]
          | (2u << 30);                                    // type=2 ("image")
  i32x8 g1;
  g1[0] = (int)((1u << 16)        // data_size=1 -> 2 bytes
              | (1u << 20)        // pad_enable
              | (3u << 22)        // pad_interval: code 3 -> every 64 bytes
              | (3u << 25));      // pad_amount:  code 3 -> 4 DWORDs = 16 bytes
  g1[1] = (int)((tensor_d0 & 0xFFFFu) << 16);              // tensor_dim0[15:0]
  g1[2] = (int)(((tensor_d0 >> 16) & 0xFFFFu)              // tensor_dim0[31:16]
              | ((tensor_d1 & 0xFFFFu) << 16));            // tensor_dim1[15:0]
  g1[3] = (int)(((tensor_d1 >> 16) & 0xFFFFu)              // tensor_dim1[31:16]
              | ((tile_d0 & 0xFFFFu) << 16));              // tile_dim0
  g1[4] = (int)(tile_d1 & 0xFFFFu);                        // tile_dim1; tile_dim2=0
  g1[5] = (int)(stride0 & 0xFFFFFFFFu);                    // dim0_stride[31:0]
  g1[6] = (int)((stride0 >> 32) & 0xFFFFu);                // dim0_stride[47:32]
  g1[7] = 0;
  i32x4 gz = {0, 0, 0, 0};                                 // dims 2+ unused
#if defined(__clang_major__) && (__clang_major__ >= 23)
  i32x8 gz8 = {0, 0, 0, 0, 0, 0, 0, 0};
  __builtin_amdgcn_tensor_load_to_lds(g0, g1, gz, gz, gz8, 0);
#else
  __builtin_amdgcn_tensor_load_to_lds(g0, g1, gz, gz, 0);
#endif
}

// =====================================================================
// Kernel 1: qkv = x @ W_qkv   (fp32 in, bf16 WMMA, f32 acc)
// 128x128 tile per WG, 256 threads = 8 waves (2 M x 4 N), K-step 32.
// Output scattered as bf16 into per-head [3][B][H][S][D] layout.
// =====================================================================
#define ASTR 40   // LDS row stride (32 + 8 pad) in bf16

__global__ __launch_bounds__(256) void gemm_qkv_kernel(
    const float* __restrict__ x, const float* __restrict__ Wq,
    bf16_t* __restrict__ qkv) {
  __shared__ __align__(16) bf16_t As[128 * ASTR];   // A tile [m][k]
  __shared__ __align__(16) bf16_t Bt[128 * ASTR];   // B tile transposed [n][k]

  const int tid  = threadIdx.x;
  const int lane = tid & 31;
  const int wave = tid >> 5;
  const int l15  = lane & 15;
  const int h    = lane >> 4;
  const int wm   = wave >> 2;   // 0..1  (64 rows each)
  const int wn   = wave & 3;    // 0..3  (32 cols each)
  const int m0   = blockIdx.y * 128;
  const int n0   = blockIdx.x * 128;

  v8f acc[4][2] = {};

  for (int kk = 0; kk < KDIM; kk += 32) {
    // ---- stage A: 128x32 fp32 -> bf16 LDS ----
#pragma unroll
    for (int i = 0; i < 4; i++) {
      int gi  = tid + i * 256;          // 0..1023 float4 groups
      int row = gi >> 3;                // 8 float4 per row
      int c4  = gi & 7;
      v4f xa = *(const v4f*)(x + (m0 + row) * KDIM + kk + c4 * 4);
      v4bf cv;
      cv[0] = (bf16_t)xa[0]; cv[1] = (bf16_t)xa[1];
      cv[2] = (bf16_t)xa[2]; cv[3] = (bf16_t)xa[3];
      *(v4bf*)&As[row * ASTR + c4 * 4] = cv;
    }
    // ---- stage B transposed: 32x128 fp32 -> bf16 LDS [n][k] ----
#pragma unroll
    for (int i = 0; i < 4; i++) {
      int gi = tid + i * 256;           // 0..1023
      int k  = gi >> 5;                 // 32 k rows
      int n4 = gi & 31;                 // 32 groups of 4 n
      v4f wv = *(const v4f*)(Wq + (kk + k) * N1 + n0 + n4 * 4);
#pragma unroll
      for (int j = 0; j < 4; j++)
        Bt[(n4 * 4 + j) * ASTR + k] = (bf16_t)wv[j];
    }
    if (kk + 32 < KDIM)
      __builtin_prefetch(Wq + (kk + 32 + (tid >> 3)) * N1 + n0, 0, 0);
    __syncthreads();

    // ---- compute: 8 WMMA per wave per K step ----
    v16bf afr[4];
#pragma unroll
    for (int a = 0; a < 4; a++)
      afr[a] = frag32(&As[(wm * 64 + a * 16 + l15) * ASTR], h);
#pragma unroll
    for (int bn = 0; bn < 2; bn++) {
      v16bf bfr = frag32(&Bt[(wn * 32 + bn * 16 + l15) * ASTR], h);
#pragma unroll
      for (int a = 0; a < 4; a++)
        acc[a][bn] = wmma_bf16(afr[a], bfr, acc[a][bn]);
    }
    __syncthreads();
  }

  // ---- epilogue: scatter bf16 into [3][B][H][S][D] ----
#pragma unroll
  for (int a = 0; a < 4; a++) {
    int Mg = m0 + wm * 64 + a * 16 + 8 * h;
#pragma unroll
    for (int bn = 0; bn < 2; bn++) {
      int Ng   = n0 + wn * 32 + bn * 16 + l15;
      int sec  = Ng >> 10;
      int rem  = Ng & 1023;
      int head = rem >> 6;
      int d    = rem & 63;
#pragma unroll
      for (int r = 0; r < 8; r++) {
        int Mr = Mg + r;
        int b  = Mr >> 11;
        int s  = Mr & 2047;
        qkv[sec * QSZ + ((b * NH + head) * SS + s) * HDIM + d] =
            (bf16_t)acc[a][bn][r];
      }
    }
  }
}

// =====================================================================
// Kernel 2: flash attention per (b, h, 64-row q tile). 4 waves x 16 rows.
// scores = 0.125 * Q K^T via WMMA; online softmax; O += P V via WMMA.
// =====================================================================
#define VSTR 72   // LDS t-stride for V^T and P (64 + 8 pad)

__global__ __launch_bounds__(128) void attn_kernel(
    const bf16_t* __restrict__ q, const bf16_t* __restrict__ k,
    const bf16_t* __restrict__ v, bf16_t* __restrict__ z) {
  __shared__ __align__(16) bf16_t vt[64 * VSTR];       // V^T tile [d][t]
  __shared__ __align__(16) bf16_t pl[4 * 16 * VSTR];   // per-wave P [m][t]

  const int tid  = threadIdx.x;
  const int lane = tid & 31;
  const int wave = tid >> 5;
  const int l15  = lane & 15;
  const int h    = lane >> 4;

  const int b    = blockIdx.z;
  const int head = blockIdx.y;
  const int qt   = blockIdx.x;
  const int base = ((b * NH + head) * SS) * HDIM;   // per-head [S][64] matrix

  // preload Q fragments (A layout): rows = this wave's 16 q rows
  const int qrow = qt * 64 + wave * 16 + l15;
  v16bf qfr[2];
#pragma unroll
  for (int kc = 0; kc < 2; kc++)
    qfr[kc] = frag32(q + base + qrow * HDIM + kc * 32, h);

  v8f Oacc[4] = {};
  float m_run[8], l_run[8];
#pragma unroll
  for (int r = 0; r < 8; r++) { m_run[r] = -3.0e38f; l_run[r] = 0.0f; }

  for (int j = 0; j < SS / 64; j++) {
    const int t0 = j * 64;
    // ---- stage V^T into LDS ----
#pragma unroll
    for (int it = 0; it < 4; it++) {
      int gi = tid + it * 128;          // 0..511 groups of 8 d
      int t  = gi >> 3;
      int d8 = (gi & 7) * 8;
      v8bf vv = *(const v8bf*)(v + base + (t0 + t) * HDIM + d8);
#pragma unroll
      for (int e = 0; e < 8; e++) vt[(d8 + e) * VSTR + t] = vv[e];
    }
    __syncthreads();

    // ---- scores = Q K^T (K fragments straight from global) ----
    v8f sc[4] = {};
#pragma unroll
    for (int tt = 0; tt < 4; tt++) {
      int tcol = t0 + tt * 16 + l15;
#pragma unroll
      for (int kc = 0; kc < 2; kc++) {
        v16bf kfr = frag32(k + base + tcol * HDIM + kc * 32, h);
        sc[tt] = wmma_bf16(qfr[kc], kfr, sc[tt]);
      }
    }

    // ---- online softmax (rows live in VGPR index, cols across lanes) ----
    float fr_[8];
#pragma unroll
    for (int tt = 0; tt < 4; tt++)
#pragma unroll
      for (int r = 0; r < 8; r++) sc[tt][r] *= 0.125f;   // 1/sqrt(64)
#pragma unroll
    for (int r = 0; r < 8; r++) {
      float mx = fmaxf(fmaxf(sc[0][r], sc[1][r]), fmaxf(sc[2][r], sc[3][r]));
      mx = fmaxf(mx, __shfl_xor(mx, 1));
      mx = fmaxf(mx, __shfl_xor(mx, 2));
      mx = fmaxf(mx, __shfl_xor(mx, 4));
      mx = fmaxf(mx, __shfl_xor(mx, 8));
      float mn = fmaxf(m_run[r], mx);
      fr_[r]   = __expf(m_run[r] - mn);
      m_run[r] = mn;
    }
#pragma unroll
    for (int r = 0; r < 8; r++) {
      float rs = 0.0f;
#pragma unroll
      for (int tt = 0; tt < 4; tt++) {
        float p = __expf(sc[tt][r] - m_run[r]);
        sc[tt][r] = p;
        rs += p;
      }
      rs += __shfl_xor(rs, 1);
      rs += __shfl_xor(rs, 2);
      rs += __shfl_xor(rs, 4);
      rs += __shfl_xor(rs, 8);
      l_run[r] = l_run[r] * fr_[r] + rs;
    }
#pragma unroll
    for (int dt = 0; dt < 4; dt++)
#pragma unroll
      for (int r = 0; r < 8; r++) Oacc[dt][r] *= fr_[r];

    // ---- transpose P: C layout -> A layout via wave-private LDS ----
    bf16_t* pw = &pl[wave * 16 * VSTR];
#pragma unroll
    for (int tt = 0; tt < 4; tt++)
#pragma unroll
      for (int r = 0; r < 8; r++)
        pw[(r + 8 * h) * VSTR + tt * 16 + l15] = (bf16_t)sc[tt][r];
    asm volatile("s_wait_dscnt 0" ::: "memory");   // wave-local LDS RAW fence

    // ---- O += P V ----
#pragma unroll
    for (int kc = 0; kc < 2; kc++) {
      v16bf pfr = frag32(&pw[l15 * VSTR + kc * 32], h);
#pragma unroll
      for (int dt = 0; dt < 4; dt++) {
        v16bf vfr = frag32(&vt[(dt * 16 + l15) * VSTR + kc * 32], h);
        Oacc[dt] = wmma_bf16(pfr, vfr, Oacc[dt]);
      }
    }
    __syncthreads();   // waves done reading vt before restaging
  }

  // ---- finalize: O /= l, store bf16 z in flat [B,H,S,D] (buggy reshape) ----
#pragma unroll
  for (int dt = 0; dt < 4; dt++) {
#pragma unroll
    for (int r = 0; r < 8; r++) {
      int row = qt * 64 + wave * 16 + r + 8 * h;
      float o = Oacc[dt][r] / l_run[r];
      z[base + row * HDIM + dt * 16 + l15] = (bf16_t)o;
    }
  }
}

// =====================================================================
// Kernel 3: out = z @ W_out   (bf16 A staged by the Tensor Data Mover,
// fp32 B converted + transposed by VALU, fp32 out)
// =====================================================================
__global__ __launch_bounds__(256) void gemm_out_kernel(
    const bf16_t* __restrict__ z, const float* __restrict__ Wo,
    float* __restrict__ out) {
  __shared__ __align__(16) bf16_t As[128 * ASTR];
  __shared__ __align__(16) bf16_t Bt[128 * ASTR];

  const int tid  = threadIdx.x;
  const int lane = tid & 31;
  const int wave = tid >> 5;
  const int l15  = lane & 15;
  const int h    = lane >> 4;
  const int wm   = wave >> 2;
  const int wn   = wave & 3;
  const int m0   = blockIdx.y * 128;
  const int n0   = blockIdx.x * 128;

  // LDS byte offset of As: flat LDS addresses carry the offset in [31:0]
  const unsigned As_off = (unsigned)(unsigned long long)(uintptr_t)&As[0];

  v8f acc[4][2] = {};

  for (int kk = 0; kk < EMB; kk += 32) {
    // ---- stage A via TDM: 128x32 bf16 tile, pad engine makes ASTR rows ----
    if (wave == 0) {
      tdm_load_tile_bf16(
          As_off,
          (unsigned long long)(uintptr_t)(z + (size_t)m0 * EMB + kk),
          /*tile_d0=*/32, /*tile_d1=*/128,
          /*tensor_d0=*/EMB, /*tensor_d1=*/M1, /*stride0=*/EMB);
    }
    // ---- stage B transposed (fp32 -> bf16), overlaps with the DMA ----
#pragma unroll
    for (int i = 0; i < 4; i++) {
      int gi = tid + i * 256;
      int k  = gi >> 5;
      int n4 = gi & 31;
      v4f wv = *(const v4f*)(Wo + (kk + k) * EMB + n0 + n4 * 4);
#pragma unroll
      for (int j = 0; j < 4; j++)
        Bt[(n4 * 4 + j) * ASTR + k] = (bf16_t)wv[j];
    }
    if (kk + 32 < EMB)
      __builtin_prefetch(Wo + (kk + 32 + (tid >> 3)) * EMB + n0, 0, 0);
    if (wave == 0)
      __builtin_amdgcn_s_wait_tensorcnt(0);   // DMA done before publishing
    __syncthreads();

    v16bf afr[4];
#pragma unroll
    for (int a = 0; a < 4; a++)
      afr[a] = frag32(&As[(wm * 64 + a * 16 + l15) * ASTR], h);
#pragma unroll
    for (int bn = 0; bn < 2; bn++) {
      v16bf bfr = frag32(&Bt[(wn * 32 + bn * 16 + l15) * ASTR], h);
#pragma unroll
      for (int a = 0; a < 4; a++)
        acc[a][bn] = wmma_bf16(afr[a], bfr, acc[a][bn]);
    }
    __syncthreads();
  }

  // ---- epilogue: fp32 coalesced stores ----
#pragma unroll
  for (int a = 0; a < 4; a++) {
    int Mg = m0 + wm * 64 + a * 16 + 8 * h;
#pragma unroll
    for (int bn = 0; bn < 2; bn++) {
      int Ng = n0 + wn * 32 + bn * 16 + l15;
#pragma unroll
      for (int r = 0; r < 8; r++)
        out[(Mg + r) * EMB + Ng] = acc[a][bn][r];
    }
  }
}

// =====================================================================
// host launch
// =====================================================================
extern "C" void kernel_launch(void* const* d_in, const int* in_sizes, int n_in,
                              void* d_out, int out_size, void* d_ws, size_t ws_size,
                              hipStream_t stream) {
  const float* x    = (const float*)d_in[0];   // [B,S,1024]
  const float* Wqkv = (const float*)d_in[1];   // [1024,3072]
  const float* Wout = (const float*)d_in[2];   // [1024,1024]
  float* out = (float*)d_out;                  // [B,S,1024]

  bf16_t* ws  = (bf16_t*)d_ws;
  bf16_t* qkv = ws;                   // q,k,v each QSZ bf16 (48 MB total)
  bf16_t* zb  = ws + (size_t)3 * QSZ; // attention output, 16 MB

  gemm_qkv_kernel<<<dim3(N1 / 128, M1 / 128), 256, 0, stream>>>(x, Wqkv, qkv);
  attn_kernel<<<dim3(SS / 64, NH, BB), 128, 0, stream>>>(
      qkv, qkv + QSZ, qkv + (size_t)2 * QSZ, zb);
  gemm_out_kernel<<<dim3(EMB / 128, M1 / 128), 256, 0, stream>>>(zb, Wout, out);
}
